// MoELayer_80444737454354
// MI455X (gfx1250) — matrix operand order repaired and checked by reading, not compiled, not run on hardware
//
#include <hip/hip_runtime.h>
#include <hip/hip_bf16.h>
#include <math.h>

// ---------------- problem constants ----------------
#define NTOK   8192      // B*T
#define DIN    1024      // D
#define DFF_   2048
#define NEXP   8
#define TOPK   2
#define CAP    2560      // ceil(1.25 * N * K / E)

typedef __attribute__((ext_vector_type(16))) __bf16       v16bf;
typedef __attribute__((ext_vector_type(2)))  __bf16       v2bf;
typedef __attribute__((ext_vector_type(8)))  float        v8f;
typedef __attribute__((ext_vector_type(4)))  unsigned int v4u;

union Frag16 { v16bf v; v4u q[2]; };

#if __has_builtin(__builtin_amdgcn_cvt_pk_bf16_f32)
__device__ __forceinline__ unsigned int pk2bf(float a, float b) {
    union { v2bf v; unsigned int u; } r;
    r.v = __builtin_amdgcn_cvt_pk_bf16_f32(a, b);   // v_cvt_pk_bf16_f32
    return r.u;
}
#else
__device__ __forceinline__ unsigned int pk2bf(float a, float b) {
    // fptrunc float->bfloat: native cvt if codegen supports it, else RNE expansion
    union { v2bf v; unsigned int u; } r;
    r.v[0] = (__bf16)a;
    r.v[1] = (__bf16)b;
    return r.u;
}
#endif
__device__ __forceinline__ unsigned short f2bf(float f) {
    return (unsigned short)(pk2bf(f, f) & 0xFFFFu);
}

// ================= router: logits -> softmax -> top2 =================
__global__ __launch_bounds__(256) void moe_router(const float* __restrict__ x,
                                                  const float* __restrict__ gw,
                                                  int*   __restrict__ topI,
                                                  float* __restrict__ topW,
                                                  float* __restrict__ probsAll) {
    __shared__ __align__(16) float sGw[NEXP * DIN];          // 32 KB
    int tid = threadIdx.x;
    for (int i = tid; i < NEXP * DIN; i += 256) sGw[i] = gw[i];
    __syncthreads();

    int wv = tid >> 5, lane = tid & 31;
    int t = blockIdx.x * 8 + wv;               // one wave per token
    const float4* xr4  = (const float4*)(x + (size_t)t * DIN);
    const float4* sGw4 = (const float4*)sGw;

    float acc[NEXP];
#pragma unroll
    for (int e = 0; e < NEXP; ++e) acc[e] = 0.f;
#pragma unroll
    for (int d4 = 0; d4 < DIN / 4 / 32; ++d4) {
        int d = lane + d4 * 32;
        float4 xv = xr4[d];
#pragma unroll
        for (int e = 0; e < NEXP; ++e) {
            float4 g = sGw4[e * (DIN / 4) + d];
            acc[e] += xv.x * g.x + xv.y * g.y + xv.z * g.z + xv.w * g.w;
        }
    }
#pragma unroll
    for (int e = 0; e < NEXP; ++e)
#pragma unroll
        for (int off = 16; off > 0; off >>= 1)
            acc[e] += __shfl_xor(acc[e], off, 32);

    if (lane == 0) {
        float mx = acc[0];
#pragma unroll
        for (int e = 1; e < NEXP; ++e) mx = fmaxf(mx, acc[e]);
        float p[NEXP], s = 0.f;
#pragma unroll
        for (int e = 0; e < NEXP; ++e) { p[e] = expf(acc[e] - mx); s += p[e]; }
        float inv = 1.f / s;
#pragma unroll
        for (int e = 0; e < NEXP; ++e) { p[e] *= inv; probsAll[t * NEXP + e] = p[e]; }
        // top-2, lowest index wins ties (matches jax.lax.top_k)
        int i0 = 0; float b0 = p[0];
#pragma unroll
        for (int e = 1; e < NEXP; ++e) if (p[e] > b0) { b0 = p[e]; i0 = e; }
        int i1 = (i0 == 0) ? 1 : 0; float b1v = p[i1];
#pragma unroll
        for (int e = 0; e < NEXP; ++e)
            if (e != i0 && p[e] > b1v) { b1v = p[e]; i1 = e; }
        float sn = 1.f / (b0 + b1v + 1e-9f);
        topI[t * 2 + 0] = i0; topI[t * 2 + 1] = i1;
        topW[t * 2 + 0] = b0 * sn; topW[t * 2 + 1] = b1v * sn;
    }
}

// ===== capacity assignment: exact slot-major token-order priority =====
__global__ void moe_assign(const int* __restrict__ topI, const float* __restrict__ topW,
                           int* __restrict__ counts, int* __restrict__ listTok,
                           float* __restrict__ listW) {
    __shared__ int cntL[32][NEXP];
    int lane = threadIdx.x;                    // 32 threads, 1 wave
    const int CHUNK = (NTOK * TOPK) / 32;      // 512 contiguous slot-major entries
    int a0 = lane * CHUNK;

    int cnt[NEXP];
#pragma unroll
    for (int e = 0; e < NEXP; ++e) cnt[e] = 0;
    for (int j = 0; j < CHUNK; ++j) {
        int a = a0 + j;
        int k = a >> 13, n = a & (NTOK - 1);   // a = k*N + n
        cnt[topI[n * 2 + k]]++;
    }
#pragma unroll
    for (int e = 0; e < NEXP; ++e) cntL[lane][e] = cnt[e];
    __syncthreads();

    int run[NEXP];
#pragma unroll
    for (int e = 0; e < NEXP; ++e) {
        int b = 0;
        for (int l = 0; l < lane; ++l) b += cntL[l][e];
        run[e] = b;
    }
    if (lane == 0) {
        for (int e = 0; e < NEXP; ++e) {
            int tot = 0;
            for (int l = 0; l < 32; ++l) tot += cntL[l][e];
            counts[e] = tot < CAP ? tot : CAP;
        }
    }
    for (int j = 0; j < CHUNK; ++j) {
        int a = a0 + j;
        int k = a >> 13, n = a & (NTOK - 1);
        int e = topI[n * 2 + k];
        int pos = run[e]++;
        if (pos < CAP) {
            listTok[e * CAP + pos] = n;
            listW[e * CAP + pos]   = topW[n * 2 + k];
        }
    }
}

// ---- shared inner-loop body: preload all frags, then 4 back-to-back WMMAs ----
__device__ __forceinline__ void wmma_step(const unsigned short* sA,
                                          const unsigned short* sB,
                                          int l16, int half, int col,
                                          v8f& c0, v8f& c1) {
    Frag16 a00, a01, a10, a11, b0, b1;
    int ak0 = half * 8,  ak1 = 32 + half * 8;
    int bk0 = half * 8,  bk1 = 32 + half * 8;
    a00.q[0] = *(const v4u*)&sA[(l16) * 64 + ak0];
    a00.q[1] = *(const v4u*)&sA[(l16) * 64 + 16 + ak0];
    a10.q[0] = *(const v4u*)&sA[(16 + l16) * 64 + ak0];
    a10.q[1] = *(const v4u*)&sA[(16 + l16) * 64 + 16 + ak0];
    a01.q[0] = *(const v4u*)&sA[(l16) * 64 + ak1];
    a01.q[1] = *(const v4u*)&sA[(l16) * 64 + 16 + ak1];
    a11.q[0] = *(const v4u*)&sA[(16 + l16) * 64 + ak1];
    a11.q[1] = *(const v4u*)&sA[(16 + l16) * 64 + 16 + ak1];
    b0.q[0]  = *(const v4u*)&sB[col * 64 + bk0];
    b0.q[1]  = *(const v4u*)&sB[col * 64 + bk0 + 16];
    b1.q[0]  = *(const v4u*)&sB[col * 64 + bk1];
    b1.q[1]  = *(const v4u*)&sB[col * 64 + bk1 + 16];
    c0 = __builtin_amdgcn_wmma_f32_16x16x32_bf16(false, a00.v, false, b0.v, (short)0, c0, false, false);
    c1 = __builtin_amdgcn_wmma_f32_16x16x32_bf16(false, a10.v, false, b0.v, (short)0, c1, false, false);
    c0 = __builtin_amdgcn_wmma_f32_16x16x32_bf16(false, a01.v, false, b1.v, (short)0, c0, false, false);
    c1 = __builtin_amdgcn_wmma_f32_16x16x32_bf16(false, a11.v, false, b1.v, (short)0, c1, false, false);
}

// ============ GEMM1: H = gelu(gather(x) @ W1[e] + b1[e]) (bf16) ============
// block tile: 32 rows x 128 cols x BK=64; 8 waves; per wave per stage: 4 WMMAs
__global__ __launch_bounds__(256) void moe_gemm1(const float* __restrict__ x,
                                                 const float* __restrict__ W1,
                                                 const float* __restrict__ b1,
                                                 const int* __restrict__ listTok,
                                                 const int* __restrict__ counts,
                                                 unsigned short* __restrict__ hBuf,
                                                 int e) {
    __shared__ __align__(16) unsigned short sA[32 * 64];    // [row][k]  4 KB
    __shared__ __align__(16) unsigned short sB[128 * 64];   // [col][k] 16 KB (transposed)
    __shared__ int sTok[32];

    int cnt = counts[e];
    int m0 = blockIdx.y * 32;
    if (m0 >= cnt) return;
    int n0 = blockIdx.x * 128;
    int tid = threadIdx.x;
    const float* W1e = W1 + (size_t)e * DIN * DFF_;

    // clamp: padded rows compute garbage that is never consumed (GEMM2 guards m<cnt)
    if (tid < 32) {
        int m = m0 + tid;
        sTok[tid] = listTok[e * CAP + (m < cnt ? m : cnt - 1)];
    }
    __syncthreads();

    int lane = tid & 31, wv = tid >> 5;
    int half = lane >> 4, l16 = lane & 15;
    int col = wv * 16 + l16;

    // loop-invariant A-gather pointers (two rows per thread)
    int rA = tid >> 4, f4 = tid & 15;
    const float* pA0 = x + (size_t)sTok[rA] * DIN + f4 * 4;
    const float* pA1 = x + (size_t)sTok[rA + 16] * DIN + f4 * 4;
    // loop-invariant B-stage source (4 k-row-pairs per thread) and dests
    int krp = tid >> 5, nc4 = tid & 31;            // kr-pair 0..7 (+8 per i), col4
    const float* pB = W1e + (size_t)(krp * 2) * DFF_ + n0 + nc4 * 4;

    v8f c0, c1;
#pragma unroll
    for (int i = 0; i < 8; ++i) { c0[i] = 0.f; c1[i] = 0.f; }

    for (int k0 = 0; k0 < DIN; k0 += 64) {
        __syncthreads();
        // stage A: 32x64 fp32 -> bf16 (gathered rows), float4 loads, uint2 stores
        {
            float4 v = *(const float4*)(pA0 + k0);
            uint2 p; p.x = pk2bf(v.x, v.y); p.y = pk2bf(v.z, v.w);
            *(uint2*)&sA[rA * 64 + f4 * 4] = p;
            v = *(const float4*)(pA1 + k0);
            p.x = pk2bf(v.x, v.y); p.y = pk2bf(v.z, v.w);
            *(uint2*)&sA[(rA + 16) * 64 + f4 * 4] = p;
        }
        // stage B: 64x128 of W1[e]; pairs of k-rows -> packed u32 stores into [col][k]
#pragma unroll
        for (int i = 0; i < 4; ++i) {
            int kr = (krp + i * 8) * 2;
            const float* src = pB + (size_t)(k0 + i * 16) * DFF_;
            float4 va = *(const float4*)src;
            float4 vb = *(const float4*)(src + DFF_);
            *(unsigned int*)&sB[(nc4 * 4 + 0) * 64 + kr] = pk2bf(va.x, vb.x);
            *(unsigned int*)&sB[(nc4 * 4 + 1) * 64 + kr] = pk2bf(va.y, vb.y);
            *(unsigned int*)&sB[(nc4 * 4 + 2) * 64 + kr] = pk2bf(va.z, vb.z);
            *(unsigned int*)&sB[(nc4 * 4 + 3) * 64 + kr] = pk2bf(va.w, vb.w);
        }
        if (k0 + 64 < DIN)
            __builtin_prefetch(&W1e[(size_t)(k0 + 64 + (tid >> 3)) * DFF_ + n0 + (tid & 7) * 16], 0, 1);
        __syncthreads();
        wmma_step(sA, sB, l16, half, col, c0, c1);
    }

    int n = n0 + col;
    float bias = b1[(size_t)e * DFF_ + n];
#pragma unroll
    for (int i = 0; i < 8; ++i) {
        int m = half * 8 + i;
        float v = c0[i] + bias;
        hBuf[(size_t)(m0 + m) * DFF_ + n] =
            f2bf(0.5f * v * (1.f + erff(v * 0.70710678118f)));
        v = c1[i] + bias;
        hBuf[(size_t)(m0 + 16 + m) * DFF_ + n] =
            f2bf(0.5f * v * (1.f + erff(v * 0.70710678118f)));
    }
}

// ===== GEMM2: out[token] += w * (H @ W2[e] + b2[e]) =====
__global__ __launch_bounds__(256) void moe_gemm2(const unsigned short* __restrict__ hBuf,
                                                 const float* __restrict__ W2,
                                                 const float* __restrict__ b2,
                                                 const int* __restrict__ listTok,
                                                 const float* __restrict__ listW,
                                                 const int* __restrict__ counts,
                                                 float* __restrict__ out, int e) {
    __shared__ __align__(16) unsigned short sA[32 * 64];
    __shared__ __align__(16) unsigned short sB[128 * 64];
    __shared__ int   sTok[32];
    __shared__ float sW[32];

    int cnt = counts[e];
    int m0 = blockIdx.y * 32;
    if (m0 >= cnt) return;
    int n0 = blockIdx.x * 128;
    int tid = threadIdx.x;
    const float* W2e = W2 + (size_t)e * DFF_ * DIN;

    if (tid < 32) {
        bool ok = (m0 + tid < cnt);
        sTok[tid] = ok ? listTok[e * CAP + m0 + tid] : 0;
        sW[tid]   = ok ? listW[e * CAP + m0 + tid]   : 0.f;
    }
    __syncthreads();

    int lane = tid & 31, wv = tid >> 5;
    int half = lane >> 4, l16 = lane & 15;
    int col = wv * 16 + l16;

    // loop-invariant A copy source (bf16 row of H) and B source pointers
    int rH = tid >> 3, qH = tid & 7;
    const unsigned short* pH = hBuf + (size_t)(m0 + rH) * DFF_ + qH * 8;
    int krp = tid >> 5, nc4 = tid & 31;
    const float* pB = W2e + (size_t)(krp * 2) * DIN + n0 + nc4 * 4;

    v8f c0, c1;
#pragma unroll
    for (int i = 0; i < 8; ++i) { c0[i] = 0.f; c1[i] = 0.f; }

    for (int k0 = 0; k0 < DFF_; k0 += 64) {
        __syncthreads();
        // stage A: 32x64 bf16 copied as uint4 (1 per thread)
        *(v4u*)&sA[rH * 64 + qH * 8] = *(const v4u*)(pH + k0);
        // stage B: 64x128 of W2[e]; k-row pairs -> packed u32 transposed stores
#pragma unroll
        for (int i = 0; i < 4; ++i) {
            int kr = (krp + i * 8) * 2;
            const float* src = pB + (size_t)(k0 + i * 16) * DIN;
            float4 va = *(const float4*)src;
            float4 vb = *(const float4*)(src + DIN);
            *(unsigned int*)&sB[(nc4 * 4 + 0) * 64 + kr] = pk2bf(va.x, vb.x);
            *(unsigned int*)&sB[(nc4 * 4 + 1) * 64 + kr] = pk2bf(va.y, vb.y);
            *(unsigned int*)&sB[(nc4 * 4 + 2) * 64 + kr] = pk2bf(va.z, vb.z);
            *(unsigned int*)&sB[(nc4 * 4 + 3) * 64 + kr] = pk2bf(va.w, vb.w);
        }
        if (k0 + 64 < DFF_)
            __builtin_prefetch(&W2e[(size_t)(k0 + 64 + (tid >> 3)) * DIN + n0 + (tid & 7) * 16], 0, 1);
        __syncthreads();
        wmma_step(sA, sB, l16, half, col, c0, c1);
    }

    int n = n0 + col;
    float bias = b2[(size_t)e * DIN + n];
#pragma unroll
    for (int i = 0; i < 8; ++i) {
        int m = half * 8 + i;
        if (m0 + m < cnt)
            atomicAdd(&out[(size_t)sTok[m] * DIN + n], (c0[i] + bias) * sW[m]);
        if (m0 + 16 + m < cnt)
            atomicAdd(&out[(size_t)sTok[16 + m] * DIN + n], (c1[i] + bias) * sW[16 + m]);
    }
}

// ================= aux loss (fixed-order, deterministic) =================
__global__ void moe_finalize(const float* __restrict__ probsAll,
                             const int* __restrict__ topI, float* __restrict__ auxOut) {
    __shared__ float sf[NEXP], sp[NEXP];
    int e = threadIdx.x;                       // blockDim = 8
    float pc = 0.f; int fc = 0;
    for (int n = 0; n < NTOK; ++n) {
        pc += probsAll[n * NEXP + e];
        fc += (topI[n * 2] == e) + (topI[n * 2 + 1] == e);
    }
    sp[e] = pc; sf[e] = (float)fc;
    __syncthreads();
    if (e == 0) {
        float s = 0.f;
        for (int i = 0; i < NEXP; ++i)
            s += (sf[i] / (float)NTOK) * (sp[i] / (float)NTOK);
        *auxOut = (float)NEXP * s;
    }
}

// ========================= host launch =========================
extern "C" void kernel_launch(void* const* d_in, const int* in_sizes, int n_in,
                              void* d_out, int out_size, void* d_ws, size_t ws_size,
                              hipStream_t stream) {
    (void)in_sizes; (void)n_in; (void)out_size; (void)ws_size;
    const float* x   = (const float*)d_in[0];
    const float* gw  = (const float*)d_in[1];
    const float* W1  = (const float*)d_in[2];
    const float* b1  = (const float*)d_in[3];
    const float* W2  = (const float*)d_in[4];
    const float* b2  = (const float*)d_in[5];
    float* out = (float*)d_out;

    char* ws = (char*)d_ws;
    size_t off = 0;
    auto alloc = [&](size_t bytes) -> void* {
        void* p = ws + off;
        off = (off + bytes + 255) & ~(size_t)255;
        return p;
    };
    int*   topI     = (int*)  alloc((size_t)NTOK * 2 * sizeof(int));
    float* topW     = (float*)alloc((size_t)NTOK * 2 * sizeof(float));
    float* probsAll = (float*)alloc((size_t)NTOK * NEXP * sizeof(float));
    int*   counts   = (int*)  alloc(NEXP * sizeof(int));
    int*   listTok  = (int*)  alloc((size_t)NEXP * CAP * sizeof(int));
    float* listW    = (float*)alloc((size_t)NEXP * CAP * sizeof(float));
    unsigned short* hBuf = (unsigned short*)alloc((size_t)CAP * DFF_ * sizeof(unsigned short));

    hipMemsetAsync(out, 0, (size_t)NTOK * DIN * sizeof(float), stream);

    moe_router<<<NTOK / 8, 256, 0, stream>>>(x, gw, topI, topW, probsAll);
    moe_assign<<<1, 32, 0, stream>>>(topI, topW, counts, listTok, listW);
    for (int e = 0; e < NEXP; ++e) {
        moe_gemm1<<<dim3(DFF_ / 128, CAP / 32), 256, 0, stream>>>(x, W1, b1, listTok,
                                                                  counts, hBuf, e);
        moe_gemm2<<<dim3(DIN / 128, CAP / 32), 256, 0, stream>>>(hBuf, W2, b2, listTok,
                                                                 listW, counts, out, e);
    }
    moe_finalize<<<1, NEXP, 0, stream>>>(probsAll, topI, out + (size_t)NTOK * DIN);
}
